// Attention_45509473468628
// MI455X (gfx1250) — matrix operand order repaired and checked by reading, not compiled, not run on hardware
//
#include <hip/hip_runtime.h>

// ---------------- types ----------------
typedef __attribute__((ext_vector_type(16))) __bf16   bf16x16;
typedef __attribute__((ext_vector_type(8)))  float    f32x8;
typedef __attribute__((ext_vector_type(4)))  float    f32x4;
typedef __attribute__((ext_vector_type(4)))  unsigned u32x4;

union Frag16 {           // 16 bf16 = one WMMA A/B operand per-lane slice
    bf16x16 v;
    u32x4   q[2];
};
union Pack8 {            // 8 bf16 = one b128 store / half A-frag
    __bf16   e[8];
    u32x4    q;
    unsigned u[4];
};

__device__ __forceinline__ f32x8 wmma_bf16(bf16x16 a, bf16x16 b, f32x8 c) {
    return __builtin_amdgcn_wmma_f32_16x16x32_bf16(
        /*neg_a=*/false, a, /*neg_b=*/false, b,
        /*c_mod=*/(short)0, c, /*reuse_a=*/false, /*reuse_b=*/false);
}

// problem sizes (fixed by the reference)
#define BB   2
#define SS   2048
#define DD   1024
#define HH   16
#define DH   64
// 1/sqrt(64) * log2(e): attention runs in exp2 domain
#define QK_SCALE (0.125f * 1.4426950408889634f)

// =====================================================================
// Kernel 1: QKV projection, transposed:  C[e][s] = (W^T)[e][d] * (x^T)[d][s]
// grid(S/128, B*H), block 256 (8 waves). Wave owns 16 tokens x 64 e for
// Q, K, V (12 accumulators).  A = W^T (bf16 LDS, packed b128 staging),
// B = x rows (fp32->bf16 registers, shared by all 12 WMMAs per k-step).
// A-frag LDS loads are 2-deep ping-ponged against the WMMA stream.
// =====================================================================
__global__ __launch_bounds__(256) void qkv_proj_kernel(
    const float* __restrict__ x,
    const float* __restrict__ Wq, const float* __restrict__ Wk,
    const float* __restrict__ Wv,
    const float* __restrict__ bq, const float* __restrict__ bk,
    const float* __restrict__ bv,
    __bf16* __restrict__ Qo, __bf16* __restrict__ Ko, __bf16* __restrict__ VTo)
{
    const int bh    = blockIdx.y;
    const int b     = bh / HH;
    const int h     = bh % HH;
    const int sbase = blockIdx.x * 128;
    const int tid   = threadIdx.x;
    const int wave  = tid >> 5;
    const int lane  = tid & 31;
    const int ln    = lane & 15;
    const int hf    = lane >> 4;
    const int kh    = hf * 8;    // A-layout K sub-base
    const int ks    = hf * 16;   // contiguous-order K base (unused here)

    __shared__ alignas(16) __bf16 WT[3][64][40]; // [mat][e][k] (+pad)

    const float* Wmat[3] = { Wq + (size_t)h * DD * DH,
                             Wk + (size_t)h * DD * DH,
                             Wv + (size_t)h * DD * DH };

    f32x8 acc[3][4];
    #pragma unroll
    for (int m = 0; m < 3; ++m)
        #pragma unroll
        for (int mt = 0; mt < 4; ++mt) acc[m][mt] = (f32x8)0.0f;

    const int srow  = sbase + wave * 16 + ln;  // token this lane feeds
    const int st_e  = tid & 63;                // staging: e row
    const int st_kq = (tid >> 6) * 8;          // staging: k quad base

#define LOAD_AF(dst, mm, mmt)                                            \
    do {                                                                 \
        const __bf16* ap_ = &WT[mm][(mmt) * 16 + ln][0];                 \
        (dst).q[0] = *(const u32x4*)(ap_ + kh);                          \
        (dst).q[1] = *(const u32x4*)(ap_ + 16 + kh);                     \
    } while (0)

    for (int k0 = 0; k0 < DD; k0 += 32) {
        __syncthreads();
        // stage W^T tiles: 8 strided f32 -> one packed b128 LDS store
        #pragma unroll
        for (int m = 0; m < 3; ++m) {
            const float* wp = Wmat[m] + (size_t)(k0 + st_kq) * DH + st_e;
            Pack8 pw;
            #pragma unroll
            for (int j = 0; j < 8; ++j) pw.e[j] = (__bf16)wp[j * DH];
            *(u32x4*)&WT[m][st_e][st_kq] = pw.q;
            if (k0 + 32 < DD) __builtin_prefetch(wp + 32 * DH, 0, 1);
        }
        __syncthreads();

        // B-frag: d-values kh..kh+7, 16+kh..16+kh+7 of this lane's token (fp32->bf16)
        const float* xr = x + ((size_t)b * SS + srow) * DD + k0 + kh;
        f32x4 g0 = *(const f32x4*)(xr);
        f32x4 g1 = *(const f32x4*)(xr + 4);
        f32x4 g2 = *(const f32x4*)(xr + 16);
        f32x4 g3 = *(const f32x4*)(xr + 20);
        bf16x16 bx;
        #pragma unroll
        for (int i = 0; i < 4; ++i) {
            bx[i]      = (__bf16)g0[i];
            bx[4 + i]  = (__bf16)g1[i];
            bx[8 + i]  = (__bf16)g2[i];
            bx[12 + i] = (__bf16)g3[i];
        }

        // 2-deep ping-pong: load frag s+1 while WMMA consumes frag s
        Frag16 fr[2];
        LOAD_AF(fr[0], 0, 0);
        #pragma unroll
        for (int s = 0; s < 12; ++s) {
            if (s < 11) LOAD_AF(fr[(s + 1) & 1], (s + 1) >> 2, (s + 1) & 3);
            acc[s >> 2][s & 3] =
                wmma_bf16(fr[s & 1].v, bx, acc[s >> 2][s & 3]);
        }
    }
#undef LOAD_AF

    // epilogue: C(m=e-local, n=token).  e = mt*16 + hf*8 + v, token = srow.
    #pragma unroll
    for (int mt = 0; mt < 4; ++mt) {
        const int e0 = mt * 16 + hf * 8;            // 8 consecutive e values
        const f32x4 bq0 = *(const f32x4*)(bq + h * DH + e0);
        const f32x4 bq1 = *(const f32x4*)(bq + h * DH + e0 + 4);
        const f32x4 bk0 = *(const f32x4*)(bk + h * DH + e0);
        const f32x4 bk1 = *(const f32x4*)(bk + h * DH + e0 + 4);
        const f32x4 bv0 = *(const f32x4*)(bv + h * DH + e0);
        const f32x4 bv1 = *(const f32x4*)(bv + h * DH + e0 + 4);

        Pack8 qo, ko;
        #pragma unroll
        for (int v = 0; v < 8; ++v) {
            const float bqv = (v < 4) ? bq0[v & 3] : bq1[v & 3];
            const float bkv = (v < 4) ? bk0[v & 3] : bk1[v & 3];
            const float bvv = (v < 4) ? bv0[v & 3] : bv1[v & 3];
            qo.e[v] = (__bf16)((acc[0][mt][v] + bqv) * QK_SCALE);
            ko.e[v] = (__bf16)(acc[1][mt][v] + bkv);
            // V^T scatter (row = e, col = token)
            VTo[((size_t)bh * DH + e0 + v) * SS + srow] =
                (__bf16)(acc[2][mt][v] + bvv);
        }
        *(u32x4*)(Qo + ((size_t)bh * SS + srow) * DH + e0) = qo.q;
        *(u32x4*)(Ko + ((size_t)bh * SS + srow) * DH + e0) = ko.q;
    }
}

// =====================================================================
// Kernel 2: causal flash attention, transposed, LDS-free, 64-key windows.
// S^T = K * Q^T  (A = K rows, B = Q rows);  Z^T = V^T * P^T.
// Lane owns query column q = qbase + ln; softmax reductions are
// in-register trees + one shfl_xor(16); P relayout = cross-half swap.
// Next window's K/V rows prefetched under the softmax VALU chain.
// grid(S/64, B*H), block 128 (4 waves, one 16-query tile each).
// =====================================================================
__global__ __launch_bounds__(128) void flash_attn_kernel(
    const __bf16* __restrict__ Q, const __bf16* __restrict__ K,
    const __bf16* __restrict__ VT, __bf16* __restrict__ Z)
{
    const int bh   = blockIdx.y;
    const int b    = bh / HH;
    const int h    = bh % HH;
    const int wave = threadIdx.x >> 5;
    const int lane = threadIdx.x & 31;
    const int ln   = lane & 15;
    const int hf   = lane >> 4;
    const int kh   = hf * 8;
    const int ks   = hf * 16;

    const int qt    = blockIdx.x * 4 + wave;
    const int qbase = qt * 16;
    const int qcol  = qbase + ln;           // this lane's query

    // B-frags: Q^T (held in registers for the whole key loop)
    const __bf16* qr = Q + ((size_t)bh * SS + qcol) * DH;
    Frag16 bq0, bq1;
    bq0.q[0] = *(const u32x4*)(qr + kh);
    bq0.q[1] = *(const u32x4*)(qr + 16 + kh);
    bq1.q[0] = *(const u32x4*)(qr + 32 + kh);
    bq1.q[1] = *(const u32x4*)(qr + 48 + kh);

    float mrun = -1e30f, lrun = 0.0f;       // per-query softmax state
    f32x8 accz[4];
    #pragma unroll
    for (int mt = 0; mt < 4; ++mt) accz[mt] = (f32x8)0.0f;

    for (int j0 = 0; j0 < qbase + 16; j0 += 64) {
        // ---- scores S^T(key, q) for 4 key tiles [j0, j0+64) ----
        f32x8 c[4];
        #pragma unroll
        for (int t = 0; t < 4; ++t) {
            const __bf16* kr = K + ((size_t)bh * SS + j0 + t * 16 + ln) * DH;
            Frag16 a0, a1;
            a0.q[0] = *(const u32x4*)(kr + kh);      a0.q[1] = *(const u32x4*)(kr + 16 + kh);
            a1.q[0] = *(const u32x4*)(kr + 32 + kh); a1.q[1] = *(const u32x4*)(kr + 48 + kh);
            c[t] = (f32x8)0.0f;
            c[t] = wmma_bf16(a0.v, bq0.v, c[t]);
            c[t] = wmma_bf16(a1.v, bq1.v, c[t]);
        }

        // prefetch next window's K/V rows under the softmax chain
        if (j0 + 64 < qbase + 16) {
            #pragma unroll
            for (int t = 0; t < 4; ++t)
                __builtin_prefetch(
                    K + ((size_t)bh * SS + j0 + 64 + t * 16 + ln) * DH, 0, 1);
            #pragma unroll
            for (int mt = 0; mt < 4; ++mt)
                __builtin_prefetch(
                    VT + ((size_t)bh * DH + mt * 16 + ln) * SS + j0 + 64, 0, 1);
        }

        // ---- causal mask (key = j0 + t*16 + hf*8 + v) ----
        #pragma unroll
        for (int t = 0; t < 4; ++t)
            #pragma unroll
            for (int v = 0; v < 8; ++v)
                if (j0 + t * 16 + kh + v > qcol) c[t][v] = -1e30f;

        // ---- online softmax over 64 keys: tree + one cross-half shfl ----
        float tm = fmaxf(fmaxf(c[0][0], c[1][0]), fmaxf(c[2][0], c[3][0]));
        #pragma unroll
        for (int v = 1; v < 8; ++v)
            tm = fmaxf(tm, fmaxf(fmaxf(c[0][v], c[1][v]),
                                 fmaxf(c[2][v], c[3][v])));
        tm = fmaxf(tm, __shfl_xor(tm, 16, 32));
        const float mn    = fmaxf(mrun, tm);
        const float alpha = __builtin_amdgcn_exp2f(mrun - mn);

        float pf[4][8];
        float rs = 0.0f;
        #pragma unroll
        for (int t = 0; t < 4; ++t)
            #pragma unroll
            for (int v = 0; v < 8; ++v) {
                pf[t][v] = __builtin_amdgcn_exp2f(c[t][v] - mn);
                rs += pf[t][v];
            }
        rs += __shfl_xor(rs, 16, 32);
        lrun = lrun * alpha + rs;
        mrun = mn;
        #pragma unroll
        for (int mt = 0; mt < 4; ++mt)
            #pragma unroll
            for (int v = 0; v < 8; ++v) accz[mt][v] *= alpha;

        // ---- P^T relayout: C layout -> B layout via cross-half swap ----
        Pack8 pk[4], ot[4];
        #pragma unroll
        for (int t = 0; t < 4; ++t)
            #pragma unroll
            for (int v = 0; v < 8; ++v) pk[t].e[v] = (__bf16)pf[t][v];
        #pragma unroll
        for (int t = 0; t < 4; ++t)
            #pragma unroll
            for (int i = 0; i < 4; ++i)
                ot[t].u[i] = (unsigned)__shfl_xor((int)pk[t].u[i], 16, 32);
        Frag16 bp0, bp1;                // rows q = ln, 32 keys each
        bp0.q[0] = hf ? ot[1].q : pk[0].q;
        bp0.q[1] = hf ? pk[1].q : ot[0].q;
        bp1.q[0] = hf ? ot[3].q : pk[2].q;
        bp1.q[1] = hf ? pk[3].q : ot[2].q;

        // ---- accz += V^T * P^T  (A = V^T rows = contiguous keys) ----
        #pragma unroll
        for (int mt = 0; mt < 4; ++mt) {
            const __bf16* vr = VT + ((size_t)bh * DH + mt * 16 + ln) * SS + j0;
            Frag16 av0, av1;
            av0.q[0] = *(const u32x4*)(vr + ks);      av0.q[1] = *(const u32x4*)(vr + ks + 8);
            av1.q[0] = *(const u32x4*)(vr + 32 + ks); av1.q[1] = *(const u32x4*)(vr + 40 + ks);
            accz[mt] = wmma_bf16(av0.v, bp0.v, accz[mt]);
            accz[mt] = wmma_bf16(av1.v, bp1.v, accz[mt]);
        }
    }

    // normalize + packed store: Z[b][s][h*64 + e], 8 consecutive e per lane
    const float rl = __builtin_amdgcn_rcpf(lrun);
    #pragma unroll
    for (int mt = 0; mt < 4; ++mt) {
        Pack8 zo;
        #pragma unroll
        for (int v = 0; v < 8; ++v) zo.e[v] = (__bf16)(accz[mt][v] * rl);
        *(u32x4*)(Z + ((size_t)b * SS + qcol) * (HH * DH)
                    + h * DH + mt * 16 + kh) = zo.q;
    }
}

// =====================================================================
// Kernel 3: out^T-formulated projection: C[d][token] = Wo^T * Z^T + b_O.
// grid(4096/128, 1024/64), block 256 (8 waves).  A = Wo^T (LDS, packed
// b128 staging, ping-ponged loads), B = Z rows (bf16, direct).
// =====================================================================
__global__ __launch_bounds__(256) void out_proj_kernel(
    const __bf16* __restrict__ Z, const float* __restrict__ Wo,
    const float* __restrict__ bO, float* __restrict__ out)
{
    const int mb   = blockIdx.x * 128;   // token base
    const int nb   = blockIdx.y * 64;    // d base
    const int tid  = threadIdx.x;
    const int wave = tid >> 5;
    const int lane = tid & 31;
    const int ln   = lane & 15;
    const int hf   = lane >> 4;
    const int kh   = hf * 8;
    const int ks   = hf * 16;

    __shared__ alignas(16) __bf16 WoT[64][40];   // WoT[d][k]

    f32x8 acc[4];
    #pragma unroll
    for (int mt = 0; mt < 4; ++mt) acc[mt] = (f32x8)0.0f;

    const int token = mb + wave * 16 + ln;
    const int st_d  = tid & 63;
    const int st_kq = (tid >> 6) * 8;

#define LOAD_WF(dst, mmt)                                                \
    do {                                                                 \
        const __bf16* ap_ = &WoT[(mmt) * 16 + ln][0];                    \
        (dst).q[0] = *(const u32x4*)(ap_ + kh);                          \
        (dst).q[1] = *(const u32x4*)(ap_ + 16 + kh);                     \
    } while (0)

    for (int k0 = 0; k0 < HH * DH; k0 += 32) {
        __syncthreads();
        {   // stage Wo^T tile: 8 strided f32 -> one packed b128 LDS store
            const float* wp = Wo + (size_t)(k0 + st_kq) * DD + nb + st_d;
            Pack8 pw;
            #pragma unroll
            for (int j = 0; j < 8; ++j) pw.e[j] = (__bf16)wp[j * DD];
            *(u32x4*)&WoT[st_d][st_kq] = pw.q;
            if (k0 + 32 < HH * DH) __builtin_prefetch(wp + 32 * DD, 0, 1);
        }
        __syncthreads();

        // B-frag: this lane's token, k = kh..kh+7, 16+kh..16+kh+7 of Z (already bf16)
        const __bf16* zr = Z + (size_t)token * (HH * DH) + k0 + kh;
        Frag16 bz;
        bz.q[0] = *(const u32x4*)(zr);
        bz.q[1] = *(const u32x4*)(zr + 16);

        Frag16 fr[2];
        LOAD_WF(fr[0], 0);
        #pragma unroll
        for (int mt = 0; mt < 4; ++mt) {
            if (mt < 3) LOAD_WF(fr[(mt + 1) & 1], mt + 1);
            acc[mt] = wmma_bf16(fr[mt & 1].v, bz.v, acc[mt]);
        }
    }
#undef LOAD_WF

    // C(m=d-local, n=token): d = nb + mt*16 + hf*8 + v  (8 consecutive)
    #pragma unroll
    for (int mt = 0; mt < 4; ++mt) {
        const int d0 = nb + mt * 16 + kh;
        const f32x4 bias0 = *(const f32x4*)(bO + d0);
        const f32x4 bias1 = *(const f32x4*)(bO + d0 + 4);
        f32x4 r0, r1;
        #pragma unroll
        for (int i = 0; i < 4; ++i) {
            r0[i] = acc[mt][i]     + bias0[i];
            r1[i] = acc[mt][4 + i] + bias1[i];
        }
        *(f32x4*)(out + (size_t)token * DD + d0)     = r0;
        *(f32x4*)(out + (size_t)token * DD + d0 + 4) = r1;
    }
}

// =====================================================================
extern "C" void kernel_launch(void* const* d_in, const int* in_sizes, int n_in,
                              void* d_out, int out_size, void* d_ws, size_t ws_size,
                              hipStream_t stream) {
    const float* x   = (const float*)d_in[0];
    const float* W_Q = (const float*)d_in[1];
    const float* W_K = (const float*)d_in[2];
    const float* W_V = (const float*)d_in[3];
    const float* W_O = (const float*)d_in[4];
    const float* b_Q = (const float*)d_in[5];
    const float* b_K = (const float*)d_in[6];
    const float* b_V = (const float*)d_in[7];
    const float* b_O = (const float*)d_in[8];
    float* out = (float*)d_out;

    const size_t NTOK = (size_t)BB * HH * SS * DH;   // 4,194,304 bf16 elems
    __bf16* Qw  = (__bf16*)d_ws;
    __bf16* Kw  = Qw  + NTOK;
    __bf16* VTw = Kw  + NTOK;
    __bf16* Zw  = VTw + NTOK;                        // total 32 MB of ws

    qkv_proj_kernel<<<dim3(SS / 128, BB * HH), 256, 0, stream>>>(
        x, W_Q, W_K, W_V, b_Q, b_K, b_V, Qw, Kw, VTw);

    flash_attn_kernel<<<dim3(SS / 64, BB * HH), 128, 0, stream>>>(
        Qw, Kw, VTw, Zw);

    out_proj_kernel<<<dim3((BB * SS) / 128, DD / 64), 256, 0, stream>>>(
        Zw, W_O, b_O, out);
}
